// XCAt_15564961480778
// MI455X (gfx1250) — matrix operand order repaired and checked by reading, not compiled, not run on hardware
//
#include <hip/hip_runtime.h>

// ---------------------------------------------------------------------------
// XCA block for MI455X (gfx1250, wave32, WMMA).
// Precision: bf16 hi/lo split ("bf16x3") on v_wmma_f32_16x16x32_bf16, f32 accum.
// Gram stage uses global_load_async_to_lds_b128 (ASYNCcnt) for K/Q staging.
// ---------------------------------------------------------------------------

typedef __attribute__((ext_vector_type(16))) __bf16 v16bf;
typedef __attribute__((ext_vector_type(8)))  float  v8f;
typedef __attribute__((ext_vector_type(4)))  unsigned int u32x4;

union FragB { v16bf bf; u32x4 u[2]; };

static constexpr int Bb = 8, Nn = 4096, Cc = 512, Hh = 8, Dh = 64;

__device__ __forceinline__ unsigned short f2bf_rn(float f) {
  unsigned u = __float_as_uint(f);
  unsigned r = u + 0x7FFFu + ((u >> 16) & 1u);
  return (unsigned short)(r >> 16);
}
__device__ __forceinline__ float bf2f(unsigned short h) {
  return __uint_as_float(((unsigned)h) << 16);
}
__device__ __forceinline__ void split_bf(float f, unsigned short& hi, unsigned short& lo) {
  hi = f2bf_rn(f);
  lo = f2bf_rn(f - bf2f(hi));
}
__device__ __forceinline__ unsigned pack2(float a, float b, unsigned& lo) {
  unsigned short ha, la, hb, lb;
  split_bf(a, ha, la);
  split_bf(b, hb, lb);
  lo = (unsigned)la | ((unsigned)lb << 16);
  return (unsigned)ha | ((unsigned)hb << 16);
}
__device__ __forceinline__ void pack8(const v8f& v, u32x4& ph, u32x4& pl) {
#pragma unroll
  for (int q = 0; q < 4; ++q) {
    unsigned lo;
    unsigned hi = pack2(v[2 * q], v[2 * q + 1], lo);
    ph[q] = hi;
    pl[q] = lo;
  }
}
__device__ __forceinline__ v8f wmma_bf(const FragB& a, const FragB& b, v8f c) {
  return __builtin_amdgcn_wmma_f32_16x16x32_bf16(false, a.bf, false, b.bf,
                                                 (short)0, c, false, false);
}
__device__ __forceinline__ float sumsq16(const FragB& h, const FragB& l) {
  float s = 0.f;
#pragma unroll
  for (int p = 0; p < 2; ++p)
#pragma unroll
    for (int q = 0; q < 4; ++q) {
      unsigned uh = h.u[p][q], ul = l.u[p][q];
      float a0 = __uint_as_float(uh << 16) + __uint_as_float(ul << 16);
      float a1 = __uint_as_float(uh & 0xffff0000u) + __uint_as_float(ul & 0xffff0000u);
      s += a0 * a0 + a1 * a1;
    }
  return s;
}

// ---- async global -> LDS (16B per lane), ASYNCcnt-tracked -------------------
__device__ __forceinline__ void async_g2l_b128(void* lds, const void* g) {
  unsigned lo = (unsigned)(unsigned long long)lds;         // LDS addr = low 32b
  unsigned long long ga = (unsigned long long)g;
  asm volatile("global_load_async_to_lds_b128 %0, %1, off"
               :: "v"(lo), "v"(ga)
               : "memory");
}
__device__ __forceinline__ void wait_async0() {
#if __has_builtin(__builtin_amdgcn_s_wait_asynccnt)
  __builtin_amdgcn_s_wait_asynccnt(0);
#else
  asm volatile("s_wait_asynccnt 0x0" ::: "memory");
#endif
}

// ---------------------------------------------------------------------------
// Generic GEMM: out[m, j] = sum_k A[m,k] * W[j,k]  (A @ W^T), bf16x3 in LDS.
// Block 256 thr = 8 waves, tile 128(M) x 64(N), K-slab 32.
// EPI 0: kv projection  -> K channel-major bf16 hi/lo, V token-major f32
// EPI 1: q  projection  -> Q channel-major bf16 hi/lo
// EPI 2: final proj     -> f32 out + bias, per-b A/W via blockIdx.z
// ---------------------------------------------------------------------------
static constexpr int LP = 28;  // LDS row pitch in dwords (56 bf16 = 112B)

template <int EPI>
__global__ __launch_bounds__(256) void gemm_bf16x3(
    const float* __restrict__ A, int lda, const float* __restrict__ W, int ldw,
    int K, unsigned short* __restrict__ outHi, unsigned short* __restrict__ outLo,
    float* __restrict__ outF, const float* __restrict__ bias) {
  __shared__ unsigned int sAhi[128 * LP], sAlo[128 * LP];
  __shared__ unsigned int sWhi[64 * LP], sWlo[64 * LP];

  const int tid = threadIdx.x;
  const int wave = tid >> 5, lane = tid & 31;
  const int l15 = lane & 15, hi16 = (lane >> 4) & 1;
  const long mBase = (long)blockIdx.x * 128;
  const int nBase = blockIdx.y * 64;

  const float* Ap = A;
  const float* Wp = W;
  if (EPI == 2) {
    Ap += (size_t)blockIdx.z * (size_t)Nn * Cc;
    Wp += (size_t)blockIdx.z * (size_t)Cc * Cc;
  }

  const int wm = (wave >> 1) * 32;
  const int wn = (wave & 1) * 32;
  const int koffA = hi16 ? 8 : 0;
  const int kbB = hi16 ? 8 : 0;

  v8f acc[2][2] = {};

  for (int kc = 0; kc < K; kc += 32) {
#pragma unroll
    for (int it = 0; it < 4; ++it) {
      int lin = tid + it * 256;
      int row = lin >> 3;
      int c4 = (lin & 7) * 4;
      const float4 f = *(const float4*)(Ap + (size_t)(mBase + row) * lda + kc + c4);
      unsigned lo0, lo1;
      unsigned h0 = pack2(f.x, f.y, lo0);
      unsigned h1 = pack2(f.z, f.w, lo1);
      int wi = row * LP + (c4 >> 1);
      sAhi[wi] = h0; sAhi[wi + 1] = h1;
      sAlo[wi] = lo0; sAlo[wi + 1] = lo1;
    }
#pragma unroll
    for (int it = 0; it < 2; ++it) {
      int lin = tid + it * 256;
      int row = lin >> 3;
      int c4 = (lin & 7) * 4;
      const float4 f = *(const float4*)(Wp + (size_t)(nBase + row) * ldw + kc + c4);
      unsigned lo0, lo1;
      unsigned h0 = pack2(f.x, f.y, lo0);
      unsigned h1 = pack2(f.z, f.w, lo1);
      int wi = row * LP + (c4 >> 1);
      sWhi[wi] = h0; sWhi[wi + 1] = h1;
      sWlo[wi] = lo0; sWlo[wi + 1] = lo1;
    }
    if (kc + 32 < K) {
      __builtin_prefetch(Ap + (size_t)(mBase + (tid >> 1)) * lda + kc + 32 + (tid & 1) * 16, 0, 1);
      if (tid < 128)
        __builtin_prefetch(Wp + (size_t)(nBase + (tid >> 1)) * ldw + kc + 32 + (tid & 1) * 16, 0, 1);
    }
    __syncthreads();

    FragB ah[2], al[2], bh[2], bl[2];
#pragma unroll
    for (int ti = 0; ti < 2; ++ti) {
      int r = (wm + ti * 16 + l15) * LP + (koffA >> 1);
      ah[ti].u[0] = *(const u32x4*)&sAhi[r];
      ah[ti].u[1] = *(const u32x4*)&sAhi[r + 8];
      al[ti].u[0] = *(const u32x4*)&sAlo[r];
      al[ti].u[1] = *(const u32x4*)&sAlo[r + 8];
    }
#pragma unroll
    for (int tj = 0; tj < 2; ++tj) {
      int r = (wn + tj * 16 + l15) * LP + (kbB >> 1);
      bh[tj].u[0] = *(const u32x4*)&sWhi[r];
      bh[tj].u[1] = *(const u32x4*)&sWhi[r + 8];
      bl[tj].u[0] = *(const u32x4*)&sWlo[r];
      bl[tj].u[1] = *(const u32x4*)&sWlo[r + 8];
    }
#pragma unroll
    for (int ti = 0; ti < 2; ++ti)
#pragma unroll
      for (int tj = 0; tj < 2; ++tj) {
        acc[ti][tj] = wmma_bf(ah[ti], bh[tj], acc[ti][tj]);
        acc[ti][tj] = wmma_bf(ah[ti], bl[tj], acc[ti][tj]);
        acc[ti][tj] = wmma_bf(al[ti], bh[tj], acc[ti][tj]);
      }
    __syncthreads();
  }

#pragma unroll
  for (int ti = 0; ti < 2; ++ti)
#pragma unroll
    for (int tj = 0; tj < 2; ++tj) {
      const long m8 = mBase + wm + ti * 16 + hi16 * 8;
      const int j = nBase + wn + tj * 16 + l15;
      if (EPI == 0) {
        const int t = j >> 9, hh = (j >> 6) & 7, d = j & 63;
        const long bb = m8 >> 12;
        const long n0 = m8 & (Nn - 1);
        if (t == 0) {
          u32x4 ph, pl;
          pack8(acc[ti][tj], ph, pl);
          size_t idx = ((size_t)(bb * Hh + hh) * Dh + d) * Nn + n0;
          *(u32x4*)(outHi + idx) = ph;
          *(u32x4*)(outLo + idx) = pl;
        } else {
#pragma unroll
          for (int i = 0; i < 8; ++i)
            outF[((size_t)(bb * Nn + n0 + i)) * Cc + hh * Dh + d] = acc[ti][tj][i];
        }
      } else if (EPI == 1) {
        const int hh = j >> 6, d = j & 63;
        const long bb = m8 >> 12;
        const long n0 = m8 & (Nn - 1);
        u32x4 ph, pl;
        pack8(acc[ti][tj], ph, pl);
        size_t idx = ((size_t)(bb * Hh + hh) * Dh + d) * Nn + n0;
        *(u32x4*)(outHi + idx) = ph;
        *(u32x4*)(outLo + idx) = pl;
      } else {
        const float bv = bias[j];
        const size_t bb = blockIdx.z;
#pragma unroll
        for (int i = 0; i < 8; ++i)
          outF[((size_t)(bb * Nn + (m8 + i))) * Cc + j] = acc[ti][tj][i] + bv;
      }
    }
}

// ---------------------------------------------------------------------------
// Per-(b,h): Gram S = Q^T K over N=4096 + q/k norms, softmax, W2 build.
// 8 waves cover disjoint (d-tile x e-half) regions over full N; Q/K chunks
// staged into LDS with async global->LDS copies (each byte read once).
// ---------------------------------------------------------------------------
static constexpr int SP = 65;  // pitch of 64x64 S/attn (f32)
static constexpr int GP = 36;  // staging pitch in dwords (72 bf16 = 144B)

__global__ __launch_bounds__(256) void gram_softmax_w2(
    const unsigned short* __restrict__ QtHi, const unsigned short* __restrict__ QtLo,
    const unsigned short* __restrict__ KtHi, const unsigned short* __restrict__ KtLo,
    const float* __restrict__ proj_w, const float* __restrict__ temperature,
    float* __restrict__ W2) {
  __shared__ float sS[64 * SP];
  __shared__ unsigned int sQh[64 * GP], sQl[64 * GP];
  __shared__ unsigned int sKh[64 * GP], sKl[64 * GP];
  __shared__ float sNq[64], sNk[64], sRk[64];

  const int bh = blockIdx.x;
  const int b = bh >> 3, h = bh & 7;
  const int tid = threadIdx.x;
  const int wave = tid >> 5, lane = tid & 31;
  const int l15 = lane & 15, hi16 = (lane >> 4) & 1;

  if (tid < 64) { sNq[tid] = 0.f; sNk[tid] = 0.f; }
  __syncthreads();

  const int dtile = wave & 3;   // 16-row d band
  const int ehalf = wave >> 2;  // 32-col e half
  const int koffA = hi16 ? 8 : 0;
  const int kbB = hi16 ? 8 : 0;
  const size_t base = (size_t)bh * Dh * Nn;

  // staging assignment: per source array, thread handles 2x 16B segments
  const int s0 = tid * 2;
  const int crow0 = s0 >> 3, cseg0 = s0 & 7;
  const int crow1 = (s0 + 1) >> 3, cseg1 = (s0 + 1) & 7;

  v8f acc[2] = {};
  float nqp = 0.f;
  float nkp[2] = {0.f, 0.f};

  for (int nc = 0; nc < Nn; nc += 64) {
    // ---- stage Q/K hi+lo [64 rows x 64 halves] chunks into LDS (async)
    async_g2l_b128(&sQh[crow0 * GP + cseg0 * 4], QtHi + base + (size_t)crow0 * Nn + nc + cseg0 * 8);
    async_g2l_b128(&sQh[crow1 * GP + cseg1 * 4], QtHi + base + (size_t)crow1 * Nn + nc + cseg1 * 8);
    async_g2l_b128(&sQl[crow0 * GP + cseg0 * 4], QtLo + base + (size_t)crow0 * Nn + nc + cseg0 * 8);
    async_g2l_b128(&sQl[crow1 * GP + cseg1 * 4], QtLo + base + (size_t)crow1 * Nn + nc + cseg1 * 8);
    async_g2l_b128(&sKh[crow0 * GP + cseg0 * 4], KtHi + base + (size_t)crow0 * Nn + nc + cseg0 * 8);
    async_g2l_b128(&sKh[crow1 * GP + cseg1 * 4], KtHi + base + (size_t)crow1 * Nn + nc + cseg1 * 8);
    async_g2l_b128(&sKl[crow0 * GP + cseg0 * 4], KtLo + base + (size_t)crow0 * Nn + nc + cseg0 * 8);
    async_g2l_b128(&sKl[crow1 * GP + cseg1 * 4], KtLo + base + (size_t)crow1 * Nn + nc + cseg1 * 8);
    wait_async0();
    __syncthreads();

    // ---- consume: 2 k-sub-chunks of 32, 2 e-tiles per wave, bf16x3 WMMA
#pragma unroll
    for (int kk = 0; kk < 64; kk += 32) {
      FragB qh, ql;
      const int wq = (dtile * 16 + l15) * GP + ((kk + koffA) >> 1);
      qh.u[0] = *(const u32x4*)&sQh[wq];
      qh.u[1] = *(const u32x4*)&sQh[wq + 8];
      ql.u[0] = *(const u32x4*)&sQl[wq];
      ql.u[1] = *(const u32x4*)&sQl[wq + 8];
      if (ehalf == 0) nqp += sumsq16(qh, ql);
#pragma unroll
      for (int et = 0; et < 2; ++et) {
        const int we = ((ehalf * 2 + et) * 16 + l15) * GP + ((kk + kbB) >> 1);
        FragB kh, kl;
        kh.u[0] = *(const u32x4*)&sKh[we];
        kh.u[1] = *(const u32x4*)&sKh[we + 8];
        kl.u[0] = *(const u32x4*)&sKl[we];
        kl.u[1] = *(const u32x4*)&sKl[we + 8];
        if (dtile == 0) nkp[et] += sumsq16(kh, kl);
        acc[et] = wmma_bf(qh, kh, acc[et]);
        acc[et] = wmma_bf(qh, kl, acc[et]);
        acc[et] = wmma_bf(ql, kh, acc[et]);
      }
    }
    __syncthreads();
  }

  // ---- each wave owns a disjoint 16x32 region of S: plain stores
#pragma unroll
  for (int et = 0; et < 2; ++et)
#pragma unroll
    for (int i = 0; i < 8; ++i)
      sS[(dtile * 16 + i + hi16 * 8) * SP + (ehalf * 2 + et) * 16 + l15] = acc[et][i];
  if (ehalf == 0) atomicAdd(&sNq[dtile * 16 + l15], nqp);
  if (dtile == 0) {
    atomicAdd(&sNk[(ehalf * 2 + 0) * 16 + l15], nkp[0]);
    atomicAdd(&sNk[(ehalf * 2 + 1) * 16 + l15], nkp[1]);
  }
  __syncthreads();

  // ---- softmax over e with F.normalize eps semantics
  const float tf = temperature[h];
  if (tid < 64) sRk[tid] = 1.f / fmaxf(sqrtf(sNk[tid]), 1e-12f);
  __syncthreads();
  if (tid < 64) {
    const int d = tid;
    const float rq = tf / fmaxf(sqrtf(sNq[d]), 1e-12f);
    float mx = -3.4e38f;
    for (int e = 0; e < 64; ++e) {
      float v = sS[d * SP + e] * rq * sRk[e];
      sS[d * SP + e] = v;
      mx = fmaxf(mx, v);
    }
    float sum = 0.f;
    for (int e = 0; e < 64; ++e) {
      float v = __expf(sS[d * SP + e] - mx);
      sS[d * SP + e] = v;
      sum += v;
    }
    const float inv = 1.f / sum;
    for (int e = 0; e < 64; ++e) sS[d * SP + e] *= inv;
  }
  __syncthreads();

  // ---- W2 block [512 x 64]: proj_w[:, h*64:h*64+64] @ attn, via WMMA
  const float* pw = proj_w + h * Dh;  // row stride Cc
#pragma unroll 1
  for (int t = 0; t < 4; ++t) {
    const int mt = wave * 4 + t;
    const int rowC = mt * 16 + l15;
    v8f acw[4] = {};
#pragma unroll
    for (int kc = 0; kc < 2; ++kc) {
      FragB ph_, pl_;
#pragma unroll
      for (int p = 0; p < 2; ++p) {
        const int dbase = kc * 32 + p * 16 + koffA;
#pragma unroll
        for (int q = 0; q < 4; ++q) {
          float f0 = pw[(size_t)rowC * Cc + dbase + 2 * q];
          float f1 = pw[(size_t)rowC * Cc + dbase + 2 * q + 1];
          unsigned lo;
          ph_.u[p][q] = pack2(f0, f1, lo);
          pl_.u[p][q] = lo;
        }
      }
#pragma unroll
      for (int et = 0; et < 4; ++et) {
        FragB bh_, bl_;
        const int e = et * 16 + l15;
#pragma unroll
        for (int p = 0; p < 2; ++p)
#pragma unroll
          for (int q = 0; q < 4; ++q) {
            float f0 = sS[(kc * 32 + kbB + p * 16 + 2 * q) * SP + e];
            float f1 = sS[(kc * 32 + kbB + p * 16 + 2 * q + 1) * SP + e];
            unsigned lo;
            bh_.u[p][q] = pack2(f0, f1, lo);
            bl_.u[p][q] = lo;
          }
        acw[et] = wmma_bf(ph_, bh_, acw[et]);
        acw[et] = wmma_bf(ph_, bl_, acw[et]);
        acw[et] = wmma_bf(pl_, bh_, acw[et]);
      }
    }
#pragma unroll
    for (int et = 0; et < 4; ++et)
#pragma unroll
      for (int i = 0; i < 8; ++i) {
        const int cp = mt * 16 + i + hi16 * 8;
        const int e = et * 16 + l15;
        W2[((size_t)(b * Cc + cp)) * Cc + h * Dh + e] = acw[et][i];
      }
  }
}

// ---------------------------------------------------------------------------
extern "C" void kernel_launch(void* const* d_in, const int* in_sizes, int n_in,
                              void* d_out, int out_size, void* d_ws, size_t ws_size,
                              hipStream_t stream) {
  (void)in_sizes; (void)n_in; (void)out_size; (void)ws_size;
  const float* x      = (const float*)d_in[0];  // [B,N,C]
  const float* y      = (const float*)d_in[1];  // [B,N,256]
  const float* kv_w   = (const float*)d_in[2];  // [1024,512]
  const float* q_w    = (const float*)d_in[3];  // [512,256]
  const float* proj_w = (const float*)d_in[4];  // [512,512]
  const float* proj_b = (const float*)d_in[5];  // [512]
  const float* temp   = (const float*)d_in[6];  // [H]

  char* ws = (char*)d_ws;
  size_t off = 0;
  auto carve = [&](size_t bytes) {
    off = (off + 255) & ~(size_t)255;
    void* p = ws + off;
    off += bytes;
    return p;
  };
  const size_t qkBytes = (size_t)Bb * Hh * Dh * Nn * sizeof(unsigned short);  // 32 MiB
  unsigned short* QtHi = (unsigned short*)carve(qkBytes);
  unsigned short* QtLo = (unsigned short*)carve(qkBytes);
  unsigned short* KtHi = (unsigned short*)carve(qkBytes);
  unsigned short* KtLo = (unsigned short*)carve(qkBytes);
  float* Vc = (float*)carve((size_t)Bb * Nn * Cc * sizeof(float));  // 64 MiB
  float* W2 = (float*)carve((size_t)Bb * Cc * Cc * sizeof(float));  //  8 MiB

  dim3 blk(256);
  // kv = x @ kv_w^T : M=32768, N=1024, K=512
  gemm_bf16x3<0><<<dim3(256, 16, 1), blk, 0, stream>>>(x, Cc, kv_w, Cc, Cc,
                                                       KtHi, KtLo, Vc, nullptr);
  // q = y @ q_w^T : M=32768, N=512, K=256
  gemm_bf16x3<1><<<dim3(256, 8, 1), blk, 0, stream>>>(y, 256, q_w, 256, 256,
                                                      QtHi, QtLo, nullptr, nullptr);
  // Gram + norms + softmax + W2 build: one workgroup per (b,h)
  gram_softmax_w2<<<dim3(Bb * Hh), blk, 0, stream>>>(QtHi, QtLo, KtHi, KtLo,
                                                     proj_w, temp, W2);
  // final = Vc @ W2^T + bias : per-b, M=4096, N=512, K=512
  gemm_bf16x3<2><<<dim3(32, 8, Bb), blk, 0, stream>>>(Vc, Cc, W2, Cc, Cc,
                                                      nullptr, nullptr,
                                                      (float*)d_out, proj_b);
}
